// Decoder_57380763074768
// MI455X (gfx1250) — matrix operand order repaired and checked by reading, not compiled, not run on hardware
//
#include <hip/hip_runtime.h>

// ---------------------------------------------------------------------------
// Attention-LSTM decoder for MI455X (gfx1250), wave32 + WMMA bf16.
// Persistent-grid kernel, bf16 weights staged in workspace (all L2-resident),
// fp32 accumulation via v_wmma_f32_16x16x32_bf16.
// ---------------------------------------------------------------------------

#define NWG   16
#define NTHR  256
#define HH    1024
#define SS    2048

typedef __attribute__((ext_vector_type(16))) __bf16         v16bf;
typedef __attribute__((ext_vector_type(8)))  float          v8f;
typedef __attribute__((ext_vector_type(16))) unsigned short v16us;
typedef __attribute__((ext_vector_type(8)))  unsigned short v8us;

// ---- workspace layout (bytes, all offsets 256-aligned) ---------------------
static constexpr size_t OFF_ATTNW = 0;                                   // 2048x2048 bf16
static constexpr size_t OFF_ETB   = OFF_ATTNW + (size_t)2048*2048*2;     // E^T 1024x2048 bf16
static constexpr size_t OFF_COMBW = OFF_ETB   + (size_t)1024*2048*2;     // 1024x2048 bf16
static constexpr size_t OFF_WIH   = OFF_COMBW + (size_t)1024*2048*2;     // 2x4096x1024 bf16
static constexpr size_t OFF_WHH   = OFF_WIH   + (size_t)2*4096*1024*2;
static constexpr size_t OFF_LIN1  = OFF_WHH   + (size_t)2*4096*1024*2;   // 64x1024 bf16
static constexpr size_t OFF_OUTS  = OFF_LIN1  + (size_t)64*1024*2;       // 2048x1024 bf16
static constexpr size_t OFF_SCORES= OFF_OUTS  + (size_t)2048*1024*2;     // 2048 f32
static constexpr size_t OFF_G1TMP = OFF_SCORES+ 2048*4;                  // 4096 f32
static constexpr size_t OFF_HID   = OFF_G1TMP + 4096*4;                  // 2048x64 f32
static constexpr size_t OFF_H0F   = OFF_HID   + (size_t)2048*64*4;       // 1024 f32
static constexpr size_t OFF_XF    = OFF_H0F   + 1024*4;
static constexpr size_t OFF_C0F   = OFF_XF    + 1024*4;
static constexpr size_t OFF_C1F   = OFF_C0F   + 1024*4;
static constexpr size_t OFF_H0B   = OFF_C1F   + 1024*4;                  // [2][1024] bf16
static constexpr size_t OFF_XB    = OFF_H0B   + 2048*2;                  // [2][1024] bf16
static constexpr size_t OFF_H1B   = OFF_XB    + 2048*2;                  // [2][1024] bf16
static constexpr size_t OFF_CTXB  = OFF_H1B   + 2048*2;                  // 1024 bf16
static constexpr size_t OFF_INPB  = OFF_CTXB  + 1024*2;                  // 1024 bf16
static constexpr size_t OFF_BAR   = OFF_INPB  + 1024*2;                  // barrier
static constexpr size_t WS_NEEDED = OFF_BAR   + 256;

// ---- helpers ---------------------------------------------------------------
__device__ inline unsigned short f2bf(float f) {
  unsigned u = __float_as_uint(f);
  unsigned r = u + 0x7FFFu + ((u >> 16) & 1u);       // round-to-nearest-even
  return (unsigned short)(r >> 16);
}

__device__ inline v16bf us_to_bf(v16us u) {
  union { v16us u; v16bf b; } c; c.u = u; return c.b;
}

// A fragment: 16 rows x 32 K, bf16 weights row-major (stride ldk).
// lane<16: elems 0-15 -> K kb..kb+15
// lane>=16: elems 0-15 -> kb+16..kb+31
__device__ inline v16bf loadA(const unsigned short* W, int ldk, int rowbase,
                              int kb, int lane) {
  int row  = rowbase + (lane & 15);
  int koff = kb + ((lane & 16) ? 16 : 0);
  const unsigned short* p = W + (size_t)row * ldk + koff;
  v8us lo = *(const v8us*)p;          // k .. k+7
  v8us hi = *(const v8us*)(p + 8);    // k+8 .. k+15
  v16us u;
#pragma unroll
  for (int i = 0; i < 8; ++i) { u[i] = lo[i]; u[8 + i] = hi[i]; }
  return us_to_bf(u);
}

// B fragment as broadcast vector: all 16 columns get the same vector.
// lanes 0-15 load x[kb..kb+15], lanes 16-31 load x[kb+16..kb+31].
__device__ inline v16bf loadBvec(const unsigned short* x, int lane) {
  const unsigned short* p = x + ((lane & 16) ? 16 : 0);
  return us_to_bf(*(const v16us*)p);
}

// B fragment as real matrix: B[k][n] = Wn[n][k], Wn row-major (N x ldk).
__device__ inline v16bf loadBmat(const unsigned short* Wn, int ldk, int nbase,
                                 int kb, int lane) {
  int n = nbase + (lane & 15);
  const unsigned short* p = Wn + (size_t)n * ldk + kb + ((lane & 16) ? 16 : 0);
  return us_to_bf(*(const v16us*)p);
}

__device__ inline v8f wmma_bf16(v16bf a, v16bf b, v8f c) {
  return __builtin_amdgcn_wmma_f32_16x16x32_bf16(false, a, false, b,
                                                 (short)0, c, false, false);
}

__device__ inline float sigm(float x) { return 1.0f / (1.0f + __expf(-x)); }

__device__ inline void gsync(unsigned* bar, unsigned target) {
  __syncthreads();
  if (threadIdx.x == 0) {
    __threadfence();
    atomicAdd(bar, 1u);
    while (__hip_atomic_load(bar, __ATOMIC_RELAXED, __HIP_MEMORY_SCOPE_AGENT) < target)
      __builtin_amdgcn_s_sleep(1);
    __threadfence();
  }
  __syncthreads();
}

__device__ inline float block_max(float v, float* sR) {
  sR[threadIdx.x] = v; __syncthreads();
  for (int s = NTHR / 2; s > 0; s >>= 1) {
    if (threadIdx.x < s) sR[threadIdx.x] = fmaxf(sR[threadIdx.x], sR[threadIdx.x + s]);
    __syncthreads();
  }
  float r = sR[0]; __syncthreads(); return r;
}

__device__ inline float block_sum(float v, float* sR) {
  sR[threadIdx.x] = v; __syncthreads();
  for (int s = NTHR / 2; s > 0; s >>= 1) {
    if (threadIdx.x < s) sR[threadIdx.x] += sR[threadIdx.x + s];
    __syncthreads();
  }
  float r = sR[0]; __syncthreads(); return r;
}

// ---- prep kernels ----------------------------------------------------------
__global__ void cvt_bf16_k(const float* __restrict__ s,
                           unsigned short* __restrict__ d, int n) {
  for (int i = blockIdx.x * blockDim.x + threadIdx.x; i < n;
       i += gridDim.x * blockDim.x)
    d[i] = f2bf(s[i]);
}

__global__ void transpose_bf16_k(const float* __restrict__ E,
                                 unsigned short* __restrict__ ET) {
  // E: (S=2048, H=1024) row-major -> ET: (H, S) bf16
  const int n = HH * SS;
  for (int i = blockIdx.x * blockDim.x + threadIdx.x; i < n;
       i += gridDim.x * blockDim.x) {
    int j = i >> 11;           // H index
    int s = i & (SS - 1);      // S index
    ET[i] = f2bf(E[(size_t)s * HH + j]);
  }
}

__global__ void init_k(const float* __restrict__ hidden,
                       const float* __restrict__ cell,
                       unsigned char* ws) {
  float* h0f = (float*)(ws + OFF_H0F);
  float* xf  = (float*)(ws + OFF_XF);
  float* c0f = (float*)(ws + OFF_C0F);
  float* c1f = (float*)(ws + OFF_C1F);
  unsigned short* h0b = (unsigned short*)(ws + OFF_H0B);
  unsigned short* xb  = (unsigned short*)(ws + OFF_XB);
  unsigned short* h1b = (unsigned short*)(ws + OFF_H1B);
  unsigned* bar = (unsigned*)(ws + OFF_BAR);
  int i = blockIdx.x * blockDim.x + threadIdx.x;
  if (i == 0) *bar = 0u;
  if (i < HH) {
    float h0v = hidden[i], h1v = hidden[HH + i];
    h0f[i] = h0v; xf[i] = h1v;
    c0f[i] = cell[i]; c1f[i] = cell[HH + i];
    h0b[i] = f2bf(h0v); h0b[HH + i] = 0;
    xb[i]  = 0;         xb[HH + i]  = 0;   // x0 = zeros
    h1b[i] = f2bf(h1v); h1b[HH + i] = 0;
  }
}

// ---- main persistent kernel ------------------------------------------------
struct P {
  const float *attn_b, *comb_b, *b_ih, *b_hh, *lin1_b, *lin2_w, *lin2_b;
  float* out;
  unsigned char* ws;
};

__global__ __launch_bounds__(NTHR) void decoder_main(P p) {
  unsigned char* ws = p.ws;
  const unsigned short* attn_wb = (const unsigned short*)(ws + OFF_ATTNW);
  const unsigned short* etb     = (const unsigned short*)(ws + OFF_ETB);
  const unsigned short* comb_wb = (const unsigned short*)(ws + OFF_COMBW);
  const unsigned short* wihb    = (const unsigned short*)(ws + OFF_WIH);
  const unsigned short* whhb    = (const unsigned short*)(ws + OFF_WHH);
  const unsigned short* lin1b   = (const unsigned short*)(ws + OFF_LIN1);
  unsigned short* outsb = (unsigned short*)(ws + OFF_OUTS);
  float* scores = (float*)(ws + OFF_SCORES);
  float* g1tmp  = (float*)(ws + OFF_G1TMP);
  float* hid    = (float*)(ws + OFF_HID);
  float* h0f    = (float*)(ws + OFF_H0F);
  float* xf     = (float*)(ws + OFF_XF);
  float* c0f    = (float*)(ws + OFF_C0F);
  float* c1f    = (float*)(ws + OFF_C1F);
  unsigned short* h0b  = (unsigned short*)(ws + OFF_H0B);
  unsigned short* xb   = (unsigned short*)(ws + OFF_XB);
  unsigned short* h1b  = (unsigned short*)(ws + OFF_H1B);
  unsigned short* ctxb = (unsigned short*)(ws + OFF_CTXB);
  unsigned short* inpb = (unsigned short*)(ws + OFF_INPB);
  unsigned* bar = (unsigned*)(ws + OFF_BAR);

  __shared__ alignas(32) unsigned short sP[SS];  // unnormalized softmax, bf16
  __shared__ float sR[NTHR];

  const int tid  = threadIdx.x;
  const int gw   = (blockIdx.x * NTHR + tid) >> 5;  // global wave id 0..127
  const int lane = tid & 31;
  unsigned sync_no = 0;

  const size_t L1W = (size_t)4096 * 1024;  // layer-1 weight offset (elements)

  for (int t = 0; t < SS; ++t) {
    const int cur = t & 1, nxt = cur ^ 1;

    // ---- Phase A: attention scores = attn_w @ [x; h0] + attn_b -----------
    {
      int rb = gw * 16;  // 128 waves x 16 rows = 2048
      v8f acc = {};
      for (int kb = 0; kb < 2 * HH; kb += 32) {
        const unsigned short* xv = (kb < HH) ? (xb + cur * HH + kb)
                                             : (h0b + cur * HH + kb - HH);
        acc = wmma_bf16(loadA(attn_wb, 2 * HH, rb, kb, lane),
                        loadBvec(xv, lane), acc);
      }
      if (lane == 0 || lane == 16) {
        int off = (lane == 16) ? 8 : 0;
#pragma unroll
        for (int v = 0; v < 8; ++v) {
          int r = rb + off + v;
          scores[r] = acc[v] + p.attn_b[r];
        }
      }
    }
    gsync(bar, ++sync_no * NWG);

    // ---- Phase B: per-WG softmax into LDS, then ctx = softmax @ E --------
    {
      float lv[8]; float mx = -3.4e38f;
#pragma unroll
      for (int k = 0; k < 8; ++k) { lv[k] = scores[tid + k * NTHR]; mx = fmaxf(mx, lv[k]); }
      mx = block_max(mx, sR);
      float sum = 0.f;
#pragma unroll
      for (int k = 0; k < 8; ++k) {
        float e = __expf(lv[k] - mx);
        sP[tid + k * NTHR] = f2bf(e);   // unnormalized
        sum += e;
      }
      sum = block_sum(sum, sR);
      float rs = 1.0f / sum;
      __syncthreads();
      if (gw < 64) {                    // 64 tiles of 16 ctx rows
        int rb = gw * 16;
        v8f acc = {};
        for (int kb = 0; kb < SS; kb += 32)
          acc = wmma_bf16(loadA(etb, SS, rb, kb, lane),
                          loadBvec((const unsigned short*)sP + kb, lane), acc);
        if (lane == 0 || lane == 16) {
          int off = (lane == 16) ? 8 : 0;
#pragma unroll
          for (int v = 0; v < 8; ++v) ctxb[rb + off + v] = f2bf(acc[v] * rs);
        }
      }
    }
    gsync(bar, ++sync_no * NWG);

    // ---- Phase C: inp = relu(comb_w @ [x; ctx] + comb_b) -----------------
    if (gw < 64) {
      int rb = gw * 16;
      v8f acc = {};
      for (int kb = 0; kb < 2 * HH; kb += 32) {
        const unsigned short* xv = (kb < HH) ? (xb + cur * HH + kb)
                                             : (ctxb + kb - HH);
        acc = wmma_bf16(loadA(comb_wb, 2 * HH, rb, kb, lane),
                        loadBvec(xv, lane), acc);
      }
      if (lane == 0 || lane == 16) {
        int off = (lane == 16) ? 8 : 0;
#pragma unroll
        for (int v = 0; v < 8; ++v) {
          int r = rb + off + v;
          float x2 = acc[v] + p.comb_b[r];
          inpb[r] = f2bf(x2 > 0.f ? x2 : 0.f);
        }
      }
    }
    gsync(bar, ++sync_no * NWG);

    // ---- Phase D: waves 0-63: LSTM layer 0 full; waves 64-127: w_hh1@h1 --
    if (gw < 64) {
      int jb = gw * 16;
      v8f aI = {}, aF = {}, aG = {}, aO = {};
      for (int kb = 0; kb < HH; kb += 32) {
        v16bf bI = loadBvec(inpb + kb, lane);
        v16bf bH = loadBvec(h0b + cur * HH + kb, lane);
        aI = wmma_bf16(loadA(wihb, HH, jb, kb, lane), bI, aI);
        aI = wmma_bf16(loadA(whhb, HH, jb, kb, lane), bH, aI);
        aF = wmma_bf16(loadA(wihb, HH, HH + jb, kb, lane), bI, aF);
        aF = wmma_bf16(loadA(whhb, HH, HH + jb, kb, lane), bH, aF);
        aG = wmma_bf16(loadA(wihb, HH, 2 * HH + jb, kb, lane), bI, aG);
        aG = wmma_bf16(loadA(whhb, HH, 2 * HH + jb, kb, lane), bH, aG);
        aO = wmma_bf16(loadA(wihb, HH, 3 * HH + jb, kb, lane), bI, aO);
        aO = wmma_bf16(loadA(whhb, HH, 3 * HH + jb, kb, lane), bH, aO);
      }
      if (lane == 0 || lane == 16) {
        int off = (lane == 16) ? 8 : 0;
#pragma unroll
        for (int v = 0; v < 8; ++v) {
          int j = jb + off + v;
          float gi = aI[v] + p.b_ih[j]          + p.b_hh[j];
          float gf = aF[v] + p.b_ih[HH + j]     + p.b_hh[HH + j];
          float gg = aG[v] + p.b_ih[2 * HH + j] + p.b_hh[2 * HH + j];
          float go = aO[v] + p.b_ih[3 * HH + j] + p.b_hh[3 * HH + j];
          float iv = sigm(gi), fv = sigm(gf), gv = tanhf(gg), ov = sigm(go);
          float cn = fv * c0f[j] + iv * gv;
          float hn = ov * tanhf(cn);
          c0f[j] = cn; h0f[j] = hn;
          h0b[nxt * HH + j] = f2bf(hn);
        }
      }
    } else {
      int jb = (gw - 64) * 16;
      v8f aI = {}, aF = {}, aG = {}, aO = {};
      const unsigned short* w1 = whhb + L1W;
      for (int kb = 0; kb < HH; kb += 32) {
        v16bf bH = loadBvec(h1b + cur * HH + kb, lane);
        aI = wmma_bf16(loadA(w1, HH, jb, kb, lane), bH, aI);
        aF = wmma_bf16(loadA(w1, HH, HH + jb, kb, lane), bH, aF);
        aG = wmma_bf16(loadA(w1, HH, 2 * HH + jb, kb, lane), bH, aG);
        aO = wmma_bf16(loadA(w1, HH, 3 * HH + jb, kb, lane), bH, aO);
      }
      if (lane == 0 || lane == 16) {
        int off = (lane == 16) ? 8 : 0;
#pragma unroll
        for (int v = 0; v < 8; ++v) {
          int j = jb + off + v;
          g1tmp[j] = aI[v]; g1tmp[HH + j] = aF[v];
          g1tmp[2 * HH + j] = aG[v]; g1tmp[3 * HH + j] = aO[v];
        }
      }
    }
    gsync(bar, ++sync_no * NWG);

    // ---- Phase E: LSTM layer 1: w_ih1 @ h0_new + precomputed hh part -----
    if (gw < 64) {
      int jb = gw * 16;
      v8f aI = {}, aF = {}, aG = {}, aO = {};
      const unsigned short* w1 = wihb + L1W;
      for (int kb = 0; kb < HH; kb += 32) {
        v16bf bI = loadBvec(h0b + nxt * HH + kb, lane);
        aI = wmma_bf16(loadA(w1, HH, jb, kb, lane), bI, aI);
        aF = wmma_bf16(loadA(w1, HH, HH + jb, kb, lane), bI, aF);
        aG = wmma_bf16(loadA(w1, HH, 2 * HH + jb, kb, lane), bI, aG);
        aO = wmma_bf16(loadA(w1, HH, 3 * HH + jb, kb, lane), bI, aO);
      }
      if (lane == 0 || lane == 16) {
        int off = (lane == 16) ? 8 : 0;
#pragma unroll
        for (int v = 0; v < 8; ++v) {
          int j = jb + off + v;
          float gi = aI[v] + g1tmp[j]          + p.b_ih[4096 + j]          + p.b_hh[4096 + j];
          float gf = aF[v] + g1tmp[HH + j]     + p.b_ih[4096 + HH + j]     + p.b_hh[4096 + HH + j];
          float gg = aG[v] + g1tmp[2 * HH + j] + p.b_ih[4096 + 2 * HH + j] + p.b_hh[4096 + 2 * HH + j];
          float go = aO[v] + g1tmp[3 * HH + j] + p.b_ih[4096 + 3 * HH + j] + p.b_hh[4096 + 3 * HH + j];
          float iv = sigm(gi), fv = sigm(gf), gv = tanhf(gg), ov = sigm(go);
          float cn = fv * c1f[j] + iv * gv;
          float hn = ov * tanhf(cn);
          c1f[j] = cn; xf[j] = hn;
          unsigned short hb = f2bf(hn);
          xb[nxt * HH + j] = hb;
          h1b[nxt * HH + j] = hb;
          outsb[(size_t)t * HH + j] = hb;
        }
      }
    }
    gsync(bar, ++sync_no * NWG);
  }

  // ---- Final GEMM: hid = relu(outs @ lin1_w^T + lin1_b)  (2048x64x1024) ---
  for (int q = 0; q < 4; ++q) {
    int tile = gw + 128 * q;       // 512 tiles: 128 row-tiles x 4 col-tiles
    int mt = tile >> 2, nt = tile & 3;
    int mb = mt * 16, nb = nt * 16;
    v8f acc = {};
    for (int kb = 0; kb < HH; kb += 32)
      acc = wmma_bf16(loadA(outsb, HH, mb, kb, lane),
                      loadBmat(lin1b, HH, nb, kb, lane), acc);
    int n = nb + (lane & 15);
    float bn = p.lin1_b[n];
    int mo = (lane & 16) ? 8 : 0;
#pragma unroll
    for (int v = 0; v < 8; ++v) {
      int m = mb + mo + v;
      float h = acc[v] + bn;
      hid[(size_t)m * 64 + n] = h > 0.f ? h : 0.f;
    }
  }
  gsync(bar, ++sync_no * NWG);

  // ---- Head: logits = hid @ lin2_w^T + b; row softmax; state outputs ------
  int gt = blockIdx.x * NTHR + tid;
  if (gt < SS) {
    float l0 = p.lin2_b[0], l1 = p.lin2_b[1];
    for (int n = 0; n < 64; ++n) {
      float h = hid[(size_t)gt * 64 + n];
      l0 += h * p.lin2_w[n];
      l1 += h * p.lin2_w[64 + n];
    }
    float m = fmaxf(l0, l1);
    float e0 = __expf(l0 - m), e1 = __expf(l1 - m);
    float s = e0 + e1;
    p.out[gt * 2]     = e0 / s;
    p.out[gt * 2 + 1] = e1 / s;
  }
  if (gt < HH) {
    p.out[4096 + gt]        = h0f[gt];   // hT layer 0
    p.out[4096 + HH + gt]   = xf[gt];    // hT layer 1
    p.out[6144 + gt]        = c0f[gt];   // cT layer 0
    p.out[6144 + HH + gt]   = c1f[gt];   // cT layer 1
  }
}

// ---------------------------------------------------------------------------
extern "C" void kernel_launch(void* const* d_in, const int* in_sizes, int n_in,
                              void* d_out, int out_size, void* d_ws, size_t ws_size,
                              hipStream_t stream) {
  (void)in_sizes; (void)n_in; (void)out_size;
  const float* E       = (const float*)d_in[0];
  const float* hidden  = (const float*)d_in[1];
  const float* cell    = (const float*)d_in[2];
  const float* attn_w  = (const float*)d_in[3];
  const float* attn_b  = (const float*)d_in[4];
  const float* comb_w  = (const float*)d_in[5];
  const float* comb_b  = (const float*)d_in[6];
  const float* w_ih    = (const float*)d_in[7];
  const float* w_hh    = (const float*)d_in[8];
  const float* b_ih    = (const float*)d_in[9];
  const float* b_hh    = (const float*)d_in[10];
  const float* lin1_w  = (const float*)d_in[11];
  const float* lin1_b  = (const float*)d_in[12];
  const float* lin2_w  = (const float*)d_in[13];
  const float* lin2_b  = (const float*)d_in[14];
  unsigned char* ws = (unsigned char*)d_ws;
  if (ws_size < WS_NEEDED) return;

  cvt_bf16_k<<<2048, 256, 0, stream>>>(attn_w, (unsigned short*)(ws + OFF_ATTNW), 2048 * 2048);
  transpose_bf16_k<<<2048, 256, 0, stream>>>(E, (unsigned short*)(ws + OFF_ETB));
  cvt_bf16_k<<<2048, 256, 0, stream>>>(comb_w, (unsigned short*)(ws + OFF_COMBW), 1024 * 2048);
  cvt_bf16_k<<<4096, 256, 0, stream>>>(w_ih, (unsigned short*)(ws + OFF_WIH), 2 * 4096 * 1024);
  cvt_bf16_k<<<4096, 256, 0, stream>>>(w_hh, (unsigned short*)(ws + OFF_WHH), 2 * 4096 * 1024);
  cvt_bf16_k<<<64, 256, 0, stream>>>(lin1_w, (unsigned short*)(ws + OFF_LIN1), 64 * 1024);
  init_k<<<4, 256, 0, stream>>>(hidden, cell, ws);

  P p{attn_b, comb_b, b_ih, b_hh, lin1_b, lin2_w, lin2_b, (float*)d_out, ws};
  decoder_main<<<NWG, NTHR, 0, stream>>>(p);
}
